// QFD2Loss_55851754717295
// MI455X (gfx1250) — matrix operand order repaired and checked by reading, not compiled, not run on hardware
//
#include <hip/hip_runtime.h>

// ---------------------------------------------------------------------------
// QFD2 loss:  mean_b( q_b^T A q_b ),  q = D - softmax(logit),  A[j,k]=1-|j-k|/31
//
// MI455X reasoning: 256 MB of input vs 4 B output -> pure bandwidth problem
// (~11 us floor at 23.3 TB/s).  The 16x32 @ 32x32 per-tile contraction
// (~2.1 GFLOP total) is pushed through v_wmma_f32_16x16x32_f16 so the VALU
// only does softmax + a short dot, keeping us on the memory roofline.
// ---------------------------------------------------------------------------

typedef __attribute__((ext_vector_type(16))) _Float16 v16h;
typedef __attribute__((ext_vector_type(8)))  float    v8f;

#define LBINS 32
#define BROWS (1u << 20)            // 1048576 rows
#define TILES (BROWS / 16)          // 65536 16-row tiles
#define BLOCKS 2048
#define WAVES_PER_BLOCK 8
#define THREADS (WAVES_PER_BLOCK * 32)
#define TOTAL_WAVES (BLOCKS * WAVES_PER_BLOCK)   // 16384
#define TILES_PER_WAVE (TILES / TOTAL_WAVES)     // 4 (exact)

__global__ __launch_bounds__(THREADS)
void qfd2_main_kernel(const float* __restrict__ logit,
                      const float* __restrict__ Dm,
                      float* __restrict__ partial)
{
    const int tid  = threadIdx.x;
    const int lane = tid & 31;
    const int warp = tid >> 5;
    const int lm   = lane & 15;          // row-in-tile (B operand col) / A row
    const bool hi  = lane >= 16;
    const int kb   = hi ? 8 : 0;         // unused; A operand uses K = j + kh
    const int kh   = hi ? 16 : 0;        // B-operand K half (row element offset)

    // --- Constant A operands (two 16x32 f16 halves of the 32x32 ordinal matrix)
    // A operand, contiguous K order used here (B matches): lane<16 holds K = 0..15; lane>=16 holds
    // K = 16..31; element j maps to K = j + kh.
    v16h a_top, a_bot;
#pragma unroll
    for (int j = 0; j < 16; ++j) {
        int k = j + kh;
        float dt = fabsf((float)(lm - k));
        float db = fabsf((float)(lm + 16 - k));
        a_top[j] = (_Float16)(1.0f - dt * (1.0f / 31.0f));
        a_bot[j] = (_Float16)(1.0f - db * (1.0f / 31.0f));
    }
    const v8f zacc = {};

    float acc = 0.0f;
    const unsigned gw = blockIdx.x * WAVES_PER_BLOCK + warp;

#pragma unroll 1
    for (unsigned it = 0; it < TILES_PER_WAVE; ++it) {
        const unsigned tile = gw * TILES_PER_WAVE + it;
        const unsigned row  = tile * 16 + lm;           // global row this lane owns
        const size_t base = (size_t)row * LBINS + kh;   // 16 contiguous floats
        const float4* lp = reinterpret_cast<const float4*>(logit + base);
        const float4* dp = reinterpret_cast<const float4*>(Dm + base);

        float x[16], dd[16];
#pragma unroll
        for (int v = 0; v < 4; ++v) {
            float4 t = lp[v];
            x[4*v+0] = t.x; x[4*v+1] = t.y; x[4*v+2] = t.z; x[4*v+3] = t.w;
        }
#pragma unroll
        for (int v = 0; v < 4; ++v) {
            float4 t = dp[v];
            dd[4*v+0] = t.x; dd[4*v+1] = t.y; dd[4*v+2] = t.z; dd[4*v+3] = t.w;
        }

        // --- softmax over the 32-bin row (split across the lane^16 pair)
        float m = x[0];
#pragma unroll
        for (int j = 1; j < 16; ++j) m = fmaxf(m, x[j]);
        m = fmaxf(m, __shfl_xor(m, 16, 32));

        float e[16], s = 0.0f;
#pragma unroll
        for (int j = 0; j < 16; ++j) { e[j] = __expf(x[j] - m); s += e[j]; }
        s += __shfl_xor(s, 16, 32);
        const float inv = __frcp_rn(s);

        // q = D - softmax ; these are Q[row][kh + j]
        float q[16];
        v16h bq;
#pragma unroll
        for (int j = 0; j < 16; ++j) {
            q[j] = fmaf(-e[j], inv, dd[j]);
            bq[j] = (_Float16)q[j];     // B-layout: element j -> K = kh + j, N = lm
        }

        // --- G^T = A_half(16x32) @ Q^T(32x16); D layout gives lane n -> G[n][i]
        v8f g1 = __builtin_amdgcn_wmma_f32_16x16x32_f16(
                     false, a_top, false, bq, (short)0, zacc, false, false);
        v8f g2 = __builtin_amdgcn_wmma_f32_16x16x32_f16(
                     false, a_bot, false, bq, (short)0, zacc, false, false);

        // --- exchange the 8 Q values the partner lane owns
        // lane<16 needs q_row[16..23] (partner's q[0..7]);
        // lane>=16 needs q_row[8..15] (partner's q[8..15]).
        float r[8];
#pragma unroll
        for (int j = 0; j < 8; ++j) {
            float send = hi ? q[j] : q[8 + j];
            r[j] = __shfl_xor(send, 16, 32);
        }

        // --- t = sum_i G[n][i] * q_row[i] for this lane's 16 i-values
        float t = 0.0f;
#pragma unroll
        for (int j = 0; j < 8; ++j) {
            float f1 = hi ? r[j]     : q[j];      // i = j (+8 if hi)
            float f2 = hi ? q[8 + j] : r[j];      // i = 16 + j (+8 if hi)
            t = fmaf(g1[j], f1, t);
            t = fmaf(g2[j], f2, t);
        }
        acc += t;
    }

    // --- fixed-order wave reduction (deterministic)
#pragma unroll
    for (int ofs = 16; ofs >= 1; ofs >>= 1)
        acc += __shfl_xor(acc, ofs, 32);

    __shared__ float wsum[WAVES_PER_BLOCK];
    if (lane == 0) wsum[warp] = acc;
    __syncthreads();
    if (tid == 0) {
        float s = 0.0f;
#pragma unroll
        for (int w = 0; w < WAVES_PER_BLOCK; ++w) s += wsum[w];
        partial[blockIdx.x] = s;
    }
}

__global__ __launch_bounds__(256)
void qfd2_final_kernel(const float* __restrict__ partial, float* __restrict__ out)
{
    float s = 0.0f;
    for (int i = threadIdx.x; i < BLOCKS; i += 256) s += partial[i];
#pragma unroll
    for (int ofs = 16; ofs >= 1; ofs >>= 1)
        s += __shfl_xor(s, ofs, 32);

    __shared__ float wsum[8];
    const int lane = threadIdx.x & 31;
    const int warp = threadIdx.x >> 5;
    if (lane == 0) wsum[warp] = s;
    __syncthreads();
    if (threadIdx.x == 0) {
        float tot = 0.0f;
#pragma unroll
        for (int w = 0; w < 8; ++w) tot += wsum[w];
        out[0] = tot * (1.0f / (float)BROWS);
    }
}

extern "C" void kernel_launch(void* const* d_in, const int* in_sizes, int n_in,
                              void* d_out, int out_size, void* d_ws, size_t ws_size,
                              hipStream_t stream)
{
    const float* logit = (const float*)d_in[0];   // D_pred_logit  [B, 32] f32
    const float* Dm    = (const float*)d_in[1];   // D             [B, 32] f32
    float* out = (float*)d_out;                   // scalar f32
    float* partial = (float*)d_ws;                // BLOCKS floats (8 KB)

    qfd2_main_kernel<<<BLOCKS, THREADS, 0, stream>>>(logit, Dm, partial);
    qfd2_final_kernel<<<1, 256, 0, stream>>>(partial, out);
}
